// TransC_44478681317817
// MI455X (gfx1250) — matrix operand order repaired and checked by reading, not compiled, not run on hardware
//
#include <hip/hip_runtime.h>

// ---------------------------------------------------------------------------
// Graph TransformerConv x3 for MI455X (gfx1250, wave32, WMMA)
// - GEMMs: bf16 in / f32 accumulate v_wmma_f32_16x16x32_bf16,
//   5 M-tiles per wave sharing B registers (software-pipelined b128 loads).
// - EK (edge key/value projection) kept in bf16; the attention-logit use of
//   EK is folded into a node-level GEMM  G = Q @ WeG  so edge_alpha reads
//   only L2-resident data.
// ---------------------------------------------------------------------------

typedef __bf16 v16bf __attribute__((ext_vector_type(16)));
typedef __bf16 v8bf  __attribute__((ext_vector_type(8)));
typedef float  v8f   __attribute__((ext_vector_type(8)));

#define NN     20000   // nodes
#define NE     320000  // edges
#define DM     256     // d_model
#define NHEADS 4
#define MT     5       // M tiles per wave (20000/80=250, 320000/80=4000 exact)

// float -> bf16 round-to-nearest-even
static __device__ __forceinline__ __bf16 f2bf(float f) {
  union { float f; unsigned u; } v; v.f = f;
  unsigned r = (v.u + 0x7fffu + ((v.u >> 16) & 1u)) >> 16;
  union { unsigned short s; __bf16 b; } o; o.s = (unsigned short)r;
  return o.b;
}
static __device__ __forceinline__ float bf2f(__bf16 b) {
  union { unsigned short s; __bf16 b; } i; i.b = b;
  union { unsigned u; float f; } o; o.u = (unsigned)i.s << 16;
  return o.f;
}

// ---------------------------------------------------------------------------
// f32 -> bf16 streaming convert (n multiple of 8)
// ---------------------------------------------------------------------------
__global__ void cvt_bf16(const float* __restrict__ in, __bf16* __restrict__ out, int n) {
  int i = (blockIdx.x * blockDim.x + threadIdx.x) * 8;
  if (i >= n) return;
  float4 a = *(const float4*)(in + i);
  float4 b = *(const float4*)(in + i + 4);
  v8bf o;
  o[0] = f2bf(a.x); o[1] = f2bf(a.y); o[2] = f2bf(a.z); o[3] = f2bf(a.w);
  o[4] = f2bf(b.x); o[5] = f2bf(b.y); o[6] = f2bf(b.z); o[7] = f2bf(b.w);
  *(v8bf*)(out + i) = o;
}

// ---------------------------------------------------------------------------
// Pack W[K x 256] (f32 row-major) into per-lane WMMA B layout (bf16):
//   Wpk[((tn*ksteps+ks)*32+lane)*16+j] = W[(ks*32+hi*16+j)*256 + tn*16+lane%16]
// ---------------------------------------------------------------------------
__global__ void pack_w(const float* __restrict__ W, __bf16* __restrict__ Wpk, int K) {
  int idx = blockIdx.x * blockDim.x + threadIdx.x;
  if (idx >= K * DM) return;
  const int j    = idx & 15;
  const int l    = (idx >> 4) & 31;
  const int rest = idx >> 9;
  const int ksteps = K >> 5;
  const int ks = rest % ksteps;
  const int tn = rest / ksteps;
  const int hi = l >> 4;
  const int srcK = ks * 32 + hi * 16 + j;
  const int colc = tn * 16 + (l & 15);
  Wpk[idx] = f2bf(W[(size_t)srcK * DM + colc]);
}

// ---------------------------------------------------------------------------
// Pack WeG (256x256, per-head block-diagonal transpose of We[64 x 256]) into
// WMMA B layout:  WeG[row=h*64+c][col=h'*64+f] = (h==h') ? We[f][h*64+c] : 0
// Used for G = Q @ WeG so that q.ek == eattr.G[dst].
// ---------------------------------------------------------------------------
__global__ void pack_weg(const float* __restrict__ We, __bf16* __restrict__ Wpk) {
  int idx = blockIdx.x * blockDim.x + threadIdx.x;
  if (idx >= DM * DM) return;
  const int j    = idx & 15;
  const int l    = (idx >> 4) & 31;
  const int rest = idx >> 9;
  const int ksteps = DM >> 5;          // 8
  const int ks = rest % ksteps;
  const int tn = rest / ksteps;
  const int hi = l >> 4;
  const int srcK = ks * 32 + hi * 16 + j;     // row index (h*64+c)
  const int colc = tn * 16 + (l & 15);        // col index (h'*64+f)
  float v = 0.0f;
  if ((srcK >> 6) == (colc >> 6))             // same head block
    v = We[(size_t)(colc & 63) * DM + srcK];  // We[f][h*64+c]
  Wpk[idx] = f2bf(v);
}

// ---------------------------------------------------------------------------
// GEMM: OUT[M x 256] = Abf[M x K] @ W (+ bias), f32 output.
// grid = (M/(16*MT), 16), block = 32. MT vertical tiles share B registers.
// ---------------------------------------------------------------------------
__global__ void wmma_gemm_bf16(const __bf16* __restrict__ Abf,
                               const __bf16* __restrict__ Wpk,
                               const float* __restrict__ bias,
                               float* __restrict__ OUT,
                               int K) {
  const int lane = threadIdx.x;
  const int mrow = lane & 15;
  const int hi   = lane >> 4;
  const int tn   = blockIdx.y;
  const int m0   = blockIdx.x * (16 * MT);
  const int col  = tn * 16 + mrow;
  const int ksteps = K >> 5;
  const int aoff = hi ? 16 : 0;

  const v16bf* bp = (const v16bf*)Wpk + (size_t)tn * ksteps * 32 + lane;

  v8f acc[MT];
#pragma unroll
  for (int t = 0; t < MT; ++t) acc[t] = (v8f){};

  for (int ks = 0; ks < ksteps; ++ks) {
    const v16bf b = bp[(size_t)ks * 32];
#pragma unroll
    for (int t = 0; t < MT; ++t) {
      const int row = m0 + t * 16 + mrow;
      const __bf16* ar = Abf + (size_t)row * K + ks * 32 + aoff;
      const v8bf alo = *(const v8bf*)(ar);
      const v8bf ahi = *(const v8bf*)(ar + 8);
      const v16bf a = __builtin_shufflevector(alo, ahi,
          0, 1, 2, 3, 4, 5, 6, 7, 8, 9, 10, 11, 12, 13, 14, 15);
      acc[t] = __builtin_amdgcn_wmma_f32_16x16x32_bf16(
          false, a, false, b, (short)0, acc[t], false, false);
    }
  }

  const float bz = bias ? bias[col] : 0.0f;
#pragma unroll
  for (int t = 0; t < MT; ++t) {
#pragma unroll
    for (int r = 0; r < 8; ++r) {
      const int row = m0 + t * 16 + r + hi * 8;
      OUT[(size_t)row * DM + col] = acc[t][r] + bz;
    }
  }
}

// ---------------------------------------------------------------------------
// Same GEMM, bf16 output (used for EK: halves its HBM traffic). No bias.
// ---------------------------------------------------------------------------
__global__ void wmma_gemm_bf16_obf(const __bf16* __restrict__ Abf,
                                   const __bf16* __restrict__ Wpk,
                                   __bf16* __restrict__ OUT,
                                   int K) {
  const int lane = threadIdx.x;
  const int mrow = lane & 15;
  const int hi   = lane >> 4;
  const int tn   = blockIdx.y;
  const int m0   = blockIdx.x * (16 * MT);
  const int col  = tn * 16 + mrow;
  const int ksteps = K >> 5;
  const int aoff = hi ? 16 : 0;

  const v16bf* bp = (const v16bf*)Wpk + (size_t)tn * ksteps * 32 + lane;

  v8f acc[MT];
#pragma unroll
  for (int t = 0; t < MT; ++t) acc[t] = (v8f){};

  for (int ks = 0; ks < ksteps; ++ks) {
    const v16bf b = bp[(size_t)ks * 32];
#pragma unroll
    for (int t = 0; t < MT; ++t) {
      const int row = m0 + t * 16 + mrow;
      const __bf16* ar = Abf + (size_t)row * K + ks * 32 + aoff;
      const v8bf alo = *(const v8bf*)(ar);
      const v8bf ahi = *(const v8bf*)(ar + 8);
      const v16bf a = __builtin_shufflevector(alo, ahi,
          0, 1, 2, 3, 4, 5, 6, 7, 8, 9, 10, 11, 12, 13, 14, 15);
      acc[t] = __builtin_amdgcn_wmma_f32_16x16x32_bf16(
          false, a, false, b, (short)0, acc[t], false, false);
    }
  }

#pragma unroll
  for (int t = 0; t < MT; ++t) {
#pragma unroll
    for (int r = 0; r < 8; ++r) {
      const int row = m0 + t * 16 + r + hi * 8;
      OUT[(size_t)row * DM + col] = f2bf(acc[t][r]);
    }
  }
}

// ---------------------------------------------------------------------------
// Utility fills / degree
// ---------------------------------------------------------------------------
__global__ void fill_f32(float* p, float v, int n) {
  int i = blockIdx.x * blockDim.x + threadIdx.x;
  if (i < n) p[i] = v;
}
__global__ void fill_i32(int* p, int v, int n) {
  int i = blockIdx.x * blockDim.x + threadIdx.x;
  if (i < n) p[i] = v;
}
__global__ void deg_kernel(const int* __restrict__ dst, float* __restrict__ deg) {
  int e = blockIdx.x * blockDim.x + threadIdx.x;
  if (e < NE) atomicAdd(&deg[dst[e]], 1.0f);
}

// ---------------------------------------------------------------------------
// Edge pass A: logit[e,h] = (q[dst].k[src] + eattr[e].G[dst,h]) / 8
// One wave per edge; lane covers 8 q/k channels and 8 eattr features.
// All gathered arrays (Q, K, G) and eattr are L2-resident.
// ---------------------------------------------------------------------------
__global__ void edge_alpha(const int* __restrict__ src, const int* __restrict__ dst,
                           const float* __restrict__ Q, const float* __restrict__ Kf,
                           const float* __restrict__ G, const float* __restrict__ EATTR,
                           float* __restrict__ ALPHA, int* __restrict__ AMAX) {
  const int lane = threadIdx.x & 31;
  const int wave = threadIdx.x >> 5;
  const int e = blockIdx.x * 8 + wave;
  const int s = src[e], d = dst[e];
  const int c  = lane * 8;              // q/k channel base (head = lane>>3)
  const int h  = lane >> 3;
  const int f0 = (lane & 7) * 8;        // eattr feature base within head dot
  const float4* q4 = (const float4*)(Q     + (size_t)d * DM + c);
  const float4* k4 = (const float4*)(Kf    + (size_t)s * DM + c);
  const float4* a4 = (const float4*)(EATTR + (size_t)e * 64 + f0);
  const float4* g4 = (const float4*)(G     + (size_t)d * DM + h * 64 + f0);
  float partial = 0.0f;
#pragma unroll
  for (int j = 0; j < 2; ++j) {
    float4 q = q4[j], k = k4[j], ea = a4[j], g = g4[j];
    partial += q.x * k.x + q.y * k.y + q.z * k.z + q.w * k.w;
    partial += ea.x * g.x + ea.y * g.y + ea.z * g.z + ea.w * g.w;
  }
  partial += __shfl_xor(partial, 1);
  partial += __shfl_xor(partial, 2);
  partial += __shfl_xor(partial, 4);
  if ((lane & 7) == 0) {
    const float logit = partial * 0.125f;   // 1/sqrt(64)
    ALPHA[e * NHEADS + h] = logit;
    union { float f; int i; } u; u.f = logit;
    const int m = (u.i >= 0) ? u.i : (u.i ^ 0x7fffffff);
    atomicMax(&AMAX[d * NHEADS + h], m);
  }
}

// ---------------------------------------------------------------------------
// Edge pass B: ea = exp(alpha - amax[dst]); denom[dst] += ea
// ---------------------------------------------------------------------------
__global__ void edge_exp(const int* __restrict__ dst, float* __restrict__ ALPHA,
                         const int* __restrict__ AMAX, float* __restrict__ DEN) {
  int idx = blockIdx.x * blockDim.x + threadIdx.x;
  if (idx >= NE * NHEADS) return;
  const int e = idx >> 2, h = idx & 3;
  const int d = dst[e];
  const int enc = AMAX[d * NHEADS + h];
  union { int i; float f; } u; u.i = (enc >= 0) ? enc : (enc ^ 0x7fffffff);
  const float ea = __expf(ALPHA[idx] - u.f);
  ALPHA[idx] = ea;
  atomicAdd(&DEN[d * NHEADS + h], ea);
}

// ---------------------------------------------------------------------------
// Edge pass C: out[dst] += (ea/denom) * (v[src] + ek[e]). EK read as bf16.
// ---------------------------------------------------------------------------
__global__ void edge_scatter(const int* __restrict__ src, const int* __restrict__ dst,
                             const float* __restrict__ V, const __bf16* __restrict__ EKbf,
                             const float* __restrict__ ALPHA, const float* __restrict__ DEN,
                             float* __restrict__ OUT) {
  const int lane = threadIdx.x & 31;
  const int wave = threadIdx.x >> 5;
  const int e = blockIdx.x * 8 + wave;
  const int s = src[e], d = dst[e];
  const int h = lane >> 3;
  const float w = ALPHA[e * NHEADS + h] / DEN[d * NHEADS + h];
  const int c = lane * 8;
  const float4* v4 = (const float4*)(V + (size_t)s * DM + c);
  const v8bf ek = *(const v8bf*)(EKbf + (size_t)e * DM + c);
  float* o = OUT + (size_t)d * DM + c;
#pragma unroll
  for (int j = 0; j < 2; ++j) {
    float4 v = v4[j];
    atomicAdd(o + j * 4 + 0, w * (v.x + bf2f(ek[j * 4 + 0])));
    atomicAdd(o + j * 4 + 1, w * (v.y + bf2f(ek[j * 4 + 1])));
    atomicAdd(o + j * 4 + 2, w * (v.z + bf2f(ek[j * 4 + 2])));
    atomicAdd(o + j * 4 + 3, w * (v.w + bf2f(ek[j * 4 + 3])));
  }
}

// ---------------------------------------------------------------------------
// Node combine: mean aggr, beta gate, skip, residual + ReLU.
// ---------------------------------------------------------------------------
__global__ void node_combine(const float* __restrict__ H, const float* __restrict__ OUT,
                             const float* __restrict__ R, const float* __restrict__ DEG,
                             const float* __restrict__ Wbeta, float* __restrict__ HN) {
  const int lane = threadIdx.x & 31;
  const int wave = threadIdx.x >> 5;
  const int i = blockIdx.x * 8 + wave;
  const int c0 = lane * 8;
  const float inv_deg = 1.0f / fmaxf(DEG[i], 1.0f);

  float o[8], r[8];
  const float4* o4 = (const float4*)(OUT + (size_t)i * DM + c0);
  const float4* r4 = (const float4*)(R   + (size_t)i * DM + c0);
#pragma unroll
  for (int j = 0; j < 2; ++j) {
    float4 ov = o4[j], rv = r4[j];
    o[j * 4 + 0] = ov.x * inv_deg; o[j * 4 + 1] = ov.y * inv_deg;
    o[j * 4 + 2] = ov.z * inv_deg; o[j * 4 + 3] = ov.w * inv_deg;
    r[j * 4 + 0] = rv.x; r[j * 4 + 1] = rv.y;
    r[j * 4 + 2] = rv.z; r[j * 4 + 3] = rv.w;
  }

  float partial = 0.0f;
#pragma unroll
  for (int j = 0; j < 8; ++j) {
    const int c = c0 + j;
    partial += o[j] * Wbeta[c] + r[j] * Wbeta[DM + c] +
               (o[j] - r[j]) * Wbeta[2 * DM + c];
  }
  partial += __shfl_xor(partial, 1);
  partial += __shfl_xor(partial, 2);
  partial += __shfl_xor(partial, 4);
  partial += __shfl_xor(partial, 8);
  partial += __shfl_xor(partial, 16);
  const float b = 1.0f / (1.0f + __expf(-partial));

#pragma unroll
  for (int j = 0; j < 8; ++j) {
    const float conv = b * r[j] + (1.0f - b) * o[j];
    const float hv = H[(size_t)i * DM + c0 + j] + conv;
    HN[(size_t)i * DM + c0 + j] = fmaxf(hv, 0.0f);
  }
}

// ---------------------------------------------------------------------------
// Host orchestration (graph-capture safe: kernel launches only)
// ---------------------------------------------------------------------------
extern "C" void kernel_launch(void* const* d_in, const int* in_sizes, int n_in,
                              void* d_out, int out_size, void* d_ws, size_t ws_size,
                              hipStream_t stream) {
  (void)in_sizes; (void)n_in; (void)out_size; (void)ws_size;

  const float* x     = (const float*)d_in[0];
  const int*   ei    = (const int*)d_in[1];
  const int*   srcp  = ei;
  const int*   dstp  = ei + NE;
  const float* eattr = (const float*)d_in[2];
  const float* lpW   = (const float*)d_in[3];
  const float* lpb   = (const float*)d_in[4];

  // ---- workspace carve-up --------------------------------------------------
  char* p = (char*)d_ws;
  auto alloc_f  = [&](size_t n) { float*  r = (float*)p;  p += n * 4; return r; };
  auto alloc_i  = [&](size_t n) { int*    r = (int*)p;    p += n * 4; return r; };
  auto alloc_bf = [&](size_t n) { __bf16* r = (__bf16*)p; p += n * 2; return r; };

  float* H     = alloc_f((size_t)NN * DM);
  float* HN    = alloc_f((size_t)NN * DM);
  float* Qb    = alloc_f((size_t)NN * DM);
  float* Kb    = alloc_f((size_t)NN * DM);
  float* Vb    = alloc_f((size_t)NN * DM);
  float* Rb    = alloc_f((size_t)NN * DM);
  float* OUTb  = alloc_f((size_t)NN * DM);
  float* Gb    = alloc_f((size_t)NN * DM);
  float* ALPHA = alloc_f((size_t)NE * NHEADS);
  float* DEN   = alloc_f((size_t)NN * NHEADS);
  float* DEG   = alloc_f((size_t)NN);
  int*   AMAX  = alloc_i((size_t)NN * NHEADS);
  __bf16* Xbf  = alloc_bf((size_t)NN * 512);
  __bf16* Ebf  = alloc_bf((size_t)NE * 64);
  __bf16* Hbf  = alloc_bf((size_t)NN * DM);
  __bf16* Qbf  = alloc_bf((size_t)NN * DM);
  __bf16* EKbf = alloc_bf((size_t)NE * DM);
  __bf16* WpkLP = alloc_bf((size_t)512 * DM);
  __bf16* WpkQ  = alloc_bf((size_t)DM * DM);
  __bf16* WpkK  = alloc_bf((size_t)DM * DM);
  __bf16* WpkV  = alloc_bf((size_t)DM * DM);
  __bf16* WpkS  = alloc_bf((size_t)DM * DM);
  __bf16* WpkE  = alloc_bf((size_t)64 * DM);
  __bf16* WpkG  = alloc_bf((size_t)DM * DM);

  // ---- one-time conversions ------------------------------------------------
  cvt_bf16<<<((size_t)NN * 512 / 8 + 255) / 256, 256, 0, stream>>>(x, Xbf, NN * 512);
  cvt_bf16<<<((size_t)NE * 64 / 8 + 255) / 256, 256, 0, stream>>>(eattr, Ebf, NE * 64);
  pack_w<<<(512 * DM + 255) / 256, 256, 0, stream>>>(lpW, WpkLP, 512);

  fill_f32<<<(NN + 255) / 256, 256, 0, stream>>>(DEG, 0.0f, NN);
  deg_kernel<<<(NE + 255) / 256, 256, 0, stream>>>(dstp, DEG);

  // h = x @ lp_W + lp_b  (K = 512)
  wmma_gemm_bf16<<<dim3(NN / (16 * MT), 16), 32, 0, stream>>>(Xbf, WpkLP, lpb, H, 512);

  float* hc = H;
  float* hn = HN;
  for (int L = 0; L < 3; ++L) {
    const int pb = 5 + L * 10;
    const float* Wq    = (const float*)d_in[pb + 0];
    const float* bq    = (const float*)d_in[pb + 1];
    const float* Wk    = (const float*)d_in[pb + 2];
    const float* bk    = (const float*)d_in[pb + 3];
    const float* Wv    = (const float*)d_in[pb + 4];
    const float* bv    = (const float*)d_in[pb + 5];
    const float* We    = (const float*)d_in[pb + 6];
    const float* Wskip = (const float*)d_in[pb + 7];
    const float* bskip = (const float*)d_in[pb + 8];
    const float* Wbeta = (const float*)d_in[pb + 9];

    // pack layer weights + convert activations
    pack_w<<<(DM * DM + 255) / 256, 256, 0, stream>>>(Wq, WpkQ, DM);
    pack_w<<<(DM * DM + 255) / 256, 256, 0, stream>>>(Wk, WpkK, DM);
    pack_w<<<(DM * DM + 255) / 256, 256, 0, stream>>>(Wv, WpkV, DM);
    pack_w<<<(DM * DM + 255) / 256, 256, 0, stream>>>(Wskip, WpkS, DM);
    pack_w<<<(64 * DM + 255) / 256, 256, 0, stream>>>(We, WpkE, 64);
    pack_weg<<<(DM * DM + 255) / 256, 256, 0, stream>>>(We, WpkG);
    cvt_bf16<<<((size_t)NN * DM / 8 + 255) / 256, 256, 0, stream>>>(hc, Hbf, NN * DM);

    // node-level projections
    wmma_gemm_bf16<<<dim3(NN / (16 * MT), 16), 32, 0, stream>>>(Hbf, WpkQ, bq, Qb, DM);
    wmma_gemm_bf16<<<dim3(NN / (16 * MT), 16), 32, 0, stream>>>(Hbf, WpkK, bk, Kb, DM);
    wmma_gemm_bf16<<<dim3(NN / (16 * MT), 16), 32, 0, stream>>>(Hbf, WpkV, bv, Vb, DM);
    wmma_gemm_bf16<<<dim3(NN / (16 * MT), 16), 32, 0, stream>>>(Hbf, WpkS, bskip, Rb, DM);
    // G = Q @ WeG  (folds the q.ek term into node-level data)
    cvt_bf16<<<((size_t)NN * DM / 8 + 255) / 256, 256, 0, stream>>>(Qb, Qbf, NN * DM);
    wmma_gemm_bf16<<<dim3(NN / (16 * MT), 16), 32, 0, stream>>>(Qbf, WpkG, nullptr, Gb, DM);
    // EK = edge_attr @ We  (M = E, K = 64), bf16 output
    wmma_gemm_bf16_obf<<<dim3(NE / (16 * MT), 16), 32, 0, stream>>>(Ebf, WpkE, EKbf, 64);

    fill_i32<<<(NN * NHEADS + 255) / 256, 256, 0, stream>>>(AMAX, (int)0x80000000, NN * NHEADS);
    fill_f32<<<(NN * NHEADS + 255) / 256, 256, 0, stream>>>(DEN, 0.0f, NN * NHEADS);
    fill_f32<<<(NN * DM + 255) / 256, 256, 0, stream>>>(OUTb, 0.0f, NN * DM);

    edge_alpha<<<NE / 8, 256, 0, stream>>>(srcp, dstp, Qb, Kb, Gb, eattr, ALPHA, AMAX);
    edge_exp<<<(NE * NHEADS + 255) / 256, 256, 0, stream>>>(dstp, ALPHA, AMAX, DEN);
    edge_scatter<<<NE / 8, 256, 0, stream>>>(srcp, dstp, Vb, EKbf, ALPHA, DEN, OUTb);

    float* outp = (L == 2) ? (float*)d_out : hn;
    node_combine<<<NN / 8, 256, 0, stream>>>(hc, OUTb, Rb, DEG, Wbeta, outp);

    float* t = hc; hc = hn; hn = t;
  }
}
